// MLPPredictor_7739531067726
// MI455X (gfx1250) — matrix operand order repaired and checked by reading, not compiled, not run on hardware
//
#include <hip/hip_runtime.h>

typedef __attribute__((ext_vector_type(16))) __bf16 v16bf;
typedef __attribute__((ext_vector_type(8)))  __bf16 v8bf;
typedef __attribute__((ext_vector_type(8)))  float  v8f;

#define DFEAT   256
#define KTOT    512
#define LDA     520   // padded LDS row stride (bf16 elems): 1040B -> 4-bank shift/row
#define M_TILE  32

// ---------------------------------------------------------------------------
// Prep: W1 [512][256] f32 (row-major) -> w1t [256][512] bf16 (column-major of W1)
// so that B-matrix lane n reads contiguous K-runs.
// ---------------------------------------------------------------------------
__global__ __launch_bounds__(256) void w1_transpose_bf16_kernel(
    const float* __restrict__ W1, __bf16* __restrict__ w1t)
{
    int idx = blockIdx.x * 256 + threadIdx.x;   // 0 .. 512*256-1
    int k = idx >> 8;        // 0..511
    int n = idx & 255;       // 0..255
    w1t[n * KTOT + k] = (__bf16)W1[k * DFEAT + n];
}

// ---------------------------------------------------------------------------
// Main fused kernel: gather -> bf16 LDS tile -> WMMA GEMM (+b1, ReLU)
//                    -> fused W2 dot (+b2) -> sigmoid
// Block: 256 threads (8 waves). 32 edges per block.
// Wave w computes column tiles j = 2w, 2w+1 for both 16-row M tiles.
// ---------------------------------------------------------------------------
__global__ __launch_bounds__(256) void edge_mlp_wmma_kernel(
    const float* __restrict__ h,   const int* __restrict__ src,
    const int*   __restrict__ dst, const __bf16* __restrict__ w1t,
    const float* __restrict__ b1,  const float* __restrict__ w2,
    const float* __restrict__ b2,  float* __restrict__ out, int E)
{
    __shared__ __align__(16) __bf16 sA[M_TILE * LDA];
    __shared__ float sP[M_TILE];

    const int tid = threadIdx.x;
    const int e0  = blockIdx.x * M_TILE;

    // seed per-edge partials with b2 (second-layer bias)
    if (tid < M_TILE) sP[tid] = b2[0];

    // ---- Fill A tile: 32 rows x 512 K, K<256 from h[src], K>=256 from h[dst]
    // Each iteration: one 8-float chunk (32B global load, 16B LDS store).
    for (int idx = tid; idx < M_TILE * (KTOT / 8); idx += 256) {
        int row   = idx >> 6;        // /64 chunks per row
        int chunk = idx & 63;
        int e = e0 + row; if (e >= E) e = E - 1;
        int node = (chunk < 32) ? src[e] : dst[e];
        const float4* p = (const float4*)(h + node * DFEAT + (chunk & 31) * 8);
        float4 f0 = p[0];
        float4 f1 = p[1];
        v8bf bb;
        bb[0]=(__bf16)f0.x; bb[1]=(__bf16)f0.y; bb[2]=(__bf16)f0.z; bb[3]=(__bf16)f0.w;
        bb[4]=(__bf16)f1.x; bb[5]=(__bf16)f1.y; bb[6]=(__bf16)f1.z; bb[7]=(__bf16)f1.w;
        *(v8bf*)(&sA[row * LDA + chunk * 8]) = bb;
    }
    __syncthreads();

    const int lane = tid & 31;
    const int wave = tid >> 5;       // 0..7
    const int r    = lane & 15;      // N column within tile / M row within tile
    const int hlf  = lane >> 4;      // lane half selects K sub-block
    const int j0   = wave * 2;       // first of this wave's two 16-col tiles

    v8f acc[2][2] = {};              // [m-tile][j-tile]

    for (int kb = 0; kb < KTOT; kb += 32) {
        // B fragments: lane = column n; elems 0..7 <-> K = kb + hlf*8 + 0..7, elems 8..15 <-> K = kb + 16 + hlf*8 + 0..7
        v16bf bfrag[2];
        #pragma unroll
        for (int jt = 0; jt < 2; ++jt) {
            const __bf16* bp = w1t + ((j0 + jt) * 16 + r) * KTOT + kb + hlf * 8;
            v8bf lo = *(const v8bf*)bp;
            v8bf hi = *(const v8bf*)(bp + 16);
            bfrag[jt] = __builtin_shufflevector(lo, hi,
                0,1,2,3,4,5,6,7,8,9,10,11,12,13,14,15);
        }
        #pragma unroll
        for (int mt = 0; mt < 2; ++mt) {
            // A fragment: lane row M = r; elems 0..7 <-> K = kb + hlf*8 + 0..7,
            //             elems 8..15 <-> K = kb + 16 + hlf*8 + 0..7
            const __bf16* ap = &sA[(mt * 16 + r) * LDA + kb + hlf * 8];
            v8bf lo = *(const v8bf*)ap;
            v8bf hi = *(const v8bf*)(ap + 16);
            v16bf afrag = __builtin_shufflevector(lo, hi,
                0,1,2,3,4,5,6,7,8,9,10,11,12,13,14,15);
            acc[mt][0] = __builtin_amdgcn_wmma_f32_16x16x32_bf16(
                false, afrag, false, bfrag[0], (short)0, acc[mt][0], false, false);
            acc[mt][1] = __builtin_amdgcn_wmma_f32_16x16x32_bf16(
                false, afrag, false, bfrag[1], (short)0, acc[mt][1], false, false);
        }
    }

    // ---- Fused second layer: relu(acc + b1[n]) * w2[n], reduce over n lanes.
    // C/D layout: lane (r,hlf), VGPR v -> element (M = hlf*8+v, N = r).
    const float bias0 = b1[j0 * 16 + r];
    const float bias1 = b1[(j0 + 1) * 16 + r];
    const float w20   = w2[j0 * 16 + r];
    const float w21   = w2[(j0 + 1) * 16 + r];

    #pragma unroll
    for (int mt = 0; mt < 2; ++mt) {
        #pragma unroll
        for (int v = 0; v < 8; ++v) {
            float p = fmaxf(acc[mt][0][v] + bias0, 0.0f) * w20
                    + fmaxf(acc[mt][1][v] + bias1, 0.0f) * w21;
            p += __shfl_xor(p, 8, 32);   // reduce across the 16 N-lanes
            p += __shfl_xor(p, 4, 32);
            p += __shfl_xor(p, 2, 32);
            p += __shfl_xor(p, 1, 32);
            if (r == 0)
                atomicAdd(&sP[mt * 16 + hlf * 8 + v], p);  // ds_add_f32
        }
    }
    __syncthreads();

    if (tid < M_TILE) {
        int e = e0 + tid;
        if (e < E) out[e] = 1.0f / (1.0f + __expf(-sP[tid]));
    }
}

// ---------------------------------------------------------------------------
extern "C" void kernel_launch(void* const* d_in, const int* in_sizes, int n_in,
                              void* d_out, int out_size, void* d_ws, size_t ws_size,
                              hipStream_t stream)
{
    const float* h   = (const float*)d_in[0];
    const int*   src = (const int*)  d_in[1];
    const int*   dst = (const int*)  d_in[2];
    const float* W1  = (const float*)d_in[3];
    const float* b1  = (const float*)d_in[4];
    const float* W2  = (const float*)d_in[5];
    const float* b2  = (const float*)d_in[6];
    float* out = (float*)d_out;
    const int E = in_sizes[1];

    __bf16* w1t = (__bf16*)d_ws;     // 256*512*2 = 256 KB scratch

    w1_transpose_bf16_kernel<<<(KTOT * DFEAT) / 256, 256, 0, stream>>>(W1, w1t);
    edge_mlp_wmma_kernel<<<(E + M_TILE - 1) / M_TILE, 256, 0, stream>>>(
        h, src, dst, w1t, b1, W2, b2, out, E);
}
